// ContinuousRefiner_63462436765735
// MI455X (gfx1250) — matrix operand order repaired and checked by reading, not compiled, not run on hardware
//
#include <hip/hip_runtime.h>
#include <hip/hip_bf16.h>

// ---------------------------------------------------------------------------
// CDNA5 (gfx1250): bf16-staged dataflow, all contractions via
// v_wmma_f32_16x16x32_bf16 (wave32, f32 accumulate).  f32 -> bf16 conversion
// happens exactly once per tensor (weights up-front, activations in the
// producing kernel's epilogue), so GEMM/attention K-loops are pure
// vector-load + wmma.
// ---------------------------------------------------------------------------

typedef __attribute__((ext_vector_type(16))) __bf16          v16bf;
typedef __attribute__((ext_vector_type(8)))  __bf16          v8bf;
typedef __attribute__((ext_vector_type(16))) unsigned short  v16us;
typedef __attribute__((ext_vector_type(8)))  float           v8f;

#define D_MODEL 512
#define T_SEQ   1024
#define B_BATCH 8
#define NHEADS  8
#define HDIM    64

__device__ __forceinline__ unsigned short f32_bf16(float f) {
  unsigned u = __builtin_bit_cast(unsigned, f);
  u += 0x7FFFu + ((u >> 16) & 1u);          // round-to-nearest-even
  return (unsigned short)(u >> 16);
}

__device__ __forceinline__ v8f wmma_bf16(v16bf a, v16bf b, v8f c) {
  return __builtin_amdgcn_wmma_f32_16x16x32_bf16(
      false, a, false, b, (short)0, c, false, false);
}

// A-fragment 16x32 (row-major bf16, stride ld).  ISA 16-bit A layout:
// lane half h, row = lane&15, K = {8h+0..7, 8h+16..23} -> two contiguous
// 8-element (16B) chunks => two global_load_b128.
__device__ __forceinline__ v16bf load_frag_a_h(const __bf16* __restrict__ base,
                                               int ld, int row, int k0, int half) {
  const __bf16* rp = base + (long)row * ld + k0 + 8 * half;
  v8bf lo = *(const v8bf*)rp;
  v8bf hi = *(const v8bf*)(rp + 16);
  return __builtin_shufflevector(lo, hi, 0, 1, 2, 3, 4, 5, 6, 7,
                                 8, 9, 10, 11, 12, 13, 14, 15);
}

// B-fragment 32x16 with B[k][n] = base[n*ld + k] (W stored [N,K] row-major,
// or K^T from a [S,hd] row-major K matrix).  ISA 16-bit B layout: lane n,
// K = {8h+0..7, 8h+16..23} -> two contiguous 16B vector loads.
__device__ __forceinline__ v16bf load_frag_bT_h(const __bf16* __restrict__ base,
                                                int ld, int col, int k0, int half) {
  const __bf16* cp = base + (long)col * ld + k0 + 8 * half;
  v8bf lo = *(const v8bf*)cp;
  v8bf hi = *(const v8bf*)(cp + 16);
  return __builtin_shufflevector(lo, hi, 0, 1, 2, 3, 4, 5, 6, 7,
                                 8, 9, 10, 11, 12, 13, 14, 15);
}

__device__ __forceinline__ float gelu_exact(float x) {
  return 0.5f * x * (1.0f + erff(x * 0.70710678118654752f));
}

// ---------------------------------------------------------------------------
// f32 -> bf16 bulk conversion (8 elements / thread, b128 in, b128 out).
// ---------------------------------------------------------------------------
__global__ __launch_bounds__(256) void cvt_bf16_kernel(
    const float* __restrict__ in, __bf16* __restrict__ out, long n8) {
  long i = (long)blockIdx.x * 256 + threadIdx.x;
  if (i >= n8) return;
  const float4* p = (const float4*)in + 2 * i;
  float4 a = p[0], b = p[1];
  uint4 o;
  o.x = (unsigned)f32_bf16(a.x) | ((unsigned)f32_bf16(a.y) << 16);
  o.y = (unsigned)f32_bf16(a.z) | ((unsigned)f32_bf16(a.w) << 16);
  o.z = (unsigned)f32_bf16(b.x) | ((unsigned)f32_bf16(b.y) << 16);
  o.w = (unsigned)f32_bf16(b.z) | ((unsigned)f32_bf16(b.w) << 16);
  ((uint4*)out)[i] = o;
}

// ---------------------------------------------------------------------------
// WMMA GEMM: out = act(A[M,K] @ W[N,K]^T + bias) (+ resid).  A, W bf16.
// Output to f32 (outf) and/or bf16 (outh).  8 waves/block, wave = 16x64 tile.
// ---------------------------------------------------------------------------
__global__ __launch_bounds__(256) void gemm_wmma_kernel(
    const __bf16* __restrict__ A, int lda,
    const __bf16* __restrict__ W,           // [N,K] row-major bf16
    const float* __restrict__ bias,         // or null
    float* __restrict__ outf,               // or null
    __bf16* __restrict__ outh,              // or null
    int ldo,
    const float* __restrict__ resid, int ldr,
    int M, int N, int K, int act) {
  const int lane = threadIdx.x & 31;
  const int wave = threadIdx.x >> 5;
  const int half = lane >> 4;
  const int r    = lane & 15;
  const int m0 = blockIdx.y * 32 + (wave >> 2) * 16;
  const int n0 = blockIdx.x * 256 + (wave & 3) * 64;
  if (m0 >= M || n0 >= N) return;          // wave-uniform; EXEC stays full

  v8f acc[4];
#pragma unroll
  for (int nt = 0; nt < 4; ++nt) {
    float bv = bias ? bias[n0 + nt * 16 + r] : 0.0f;
#pragma unroll
    for (int i = 0; i < 8; ++i) acc[nt][i] = bv;
  }

  for (int k0 = 0; k0 < K; k0 += 32) {
    if (k0 + 32 < K)  // global_prefetch_b8 hint for next A block
      __builtin_prefetch(A + (long)(m0 + r) * lda + k0 + 32, 0, 1);
    v16bf af = load_frag_a_h(A, lda, m0 + r, k0, half);
#pragma unroll
    for (int nt = 0; nt < 4; ++nt) {
      v16bf bf = load_frag_bT_h(W, K, n0 + nt * 16 + r, k0, half);
      acc[nt] = wmma_bf16(af, bf, acc[nt]);
    }
  }

  // C layout: lane holds col = n0+nt*16+r; VGPR rr -> row = m0 + 8*half + rr
#pragma unroll
  for (int nt = 0; nt < 4; ++nt) {
    int col = n0 + nt * 16 + r;
#pragma unroll
    for (int rr = 0; rr < 8; ++rr) {
      long row = m0 + half * 8 + rr;
      float v = acc[nt][rr];
      if (act == 1) v = gelu_exact(v);
      if (resid) v += resid[row * ldr + col];
      if (outf) outf[row * ldo + col] = v;
      if (outh) ((unsigned short*)outh)[row * ldo + col] = f32_bf16(v);
    }
  }
}

// ---------------------------------------------------------------------------
// LayerNorm over D=512, one wave per row, bf16 output (feeds GEMM A).
// ---------------------------------------------------------------------------
__global__ __launch_bounds__(256) void ln_kernel(const float* __restrict__ x,
                                                 const float* __restrict__ g,
                                                 const float* __restrict__ b,
                                                 __bf16* __restrict__ o, int rows) {
  int row  = blockIdx.x * 8 + (threadIdx.x >> 5);
  int lane = threadIdx.x & 31;
  if (row >= rows) return;
  const float4* xr = (const float4*)(x + (long)row * D_MODEL);
  float4 v[4];
  float s = 0.f;
#pragma unroll
  for (int i = 0; i < 4; ++i) {
    v[i] = xr[lane + i * 32];
    s += v[i].x + v[i].y + v[i].z + v[i].w;
  }
#pragma unroll
  for (int m = 16; m; m >>= 1) s += __shfl_xor(s, m, 32);
  float mean = s * (1.0f / D_MODEL);
  float q = 0.f;
#pragma unroll
  for (int i = 0; i < 4; ++i) {
    float dx;
    dx = v[i].x - mean; q += dx * dx;
    dx = v[i].y - mean; q += dx * dx;
    dx = v[i].z - mean; q += dx * dx;
    dx = v[i].w - mean; q += dx * dx;
  }
#pragma unroll
  for (int m = 16; m; m >>= 1) q += __shfl_xor(q, m, 32);
  float inv = rsqrtf(q * (1.0f / D_MODEL) + 1e-5f);
  uint2* orp = (uint2*)(o + (long)row * D_MODEL);
#pragma unroll
  for (int i = 0; i < 4; ++i) {
    int e = (lane + i * 32) * 4;
    float a0 = (v[i].x - mean) * inv * g[e + 0] + b[e + 0];
    float a1 = (v[i].y - mean) * inv * g[e + 1] + b[e + 1];
    float a2 = (v[i].z - mean) * inv * g[e + 2] + b[e + 2];
    float a3 = (v[i].w - mean) * inv * g[e + 3] + b[e + 3];
    uint2 w2;
    w2.x = (unsigned)f32_bf16(a0) | ((unsigned)f32_bf16(a1) << 16);
    w2.y = (unsigned)f32_bf16(a2) | ((unsigned)f32_bf16(a3) << 16);
    orp[lane + i * 32] = w2;
  }
}

// ---------------------------------------------------------------------------
// LPF coeff generator: c = softmax(g2(silu(g1([A,H]))))[4].  One wave/token.
// ---------------------------------------------------------------------------
__global__ __launch_bounds__(256) void coeff_kernel(
    const float* __restrict__ Ag, const float* __restrict__ Hg,
    const float* __restrict__ g1w, const float* __restrict__ g1b,
    const float* __restrict__ g2w, const float* __restrict__ g2b,
    float* __restrict__ cbuf, int ntok) {
  int tok  = blockIdx.x * 8 + (threadIdx.x >> 5);
  int lane = threadIdx.x & 31;
  if (tok >= ntok) return;
  float a = Ag[tok], hh = Hg[tok];
  float logit[9];
#pragma unroll
  for (int k = 0; k < 9; ++k) logit[k] = 0.f;
  for (int j = lane; j < D_MODEL; j += 32) {
    float t  = a * g1w[2 * j] + hh * g1w[2 * j + 1] + g1b[j];
    float hj = t / (1.0f + __expf(-t));      // SiLU
#pragma unroll
    for (int k = 0; k < 9; ++k) logit[k] += hj * g2w[k * D_MODEL + j];
  }
#pragma unroll
  for (int k = 0; k < 9; ++k) {
#pragma unroll
    for (int m = 16; m; m >>= 1) logit[k] += __shfl_xor(logit[k], m, 32);
    logit[k] += g2b[k];
  }
  float mx = logit[0];
#pragma unroll
  for (int k = 1; k < 9; ++k) mx = fmaxf(mx, logit[k]);
  float se = 0.f;
#pragma unroll
  for (int k = 0; k < 9; ++k) se += __expf(logit[k] - mx);
  float c = __expf(logit[4] - mx) / se;
  if (lane == 0) cbuf[tok] = c;
}

// ---------------------------------------------------------------------------
// Depthwise 9-tap time conv + center-tap blend + residual (f32 stream).
// ---------------------------------------------------------------------------
__global__ __launch_bounds__(256) void lpf_kernel(const float* __restrict__ y,
                                                  const float* __restrict__ dw,
                                                  const float* __restrict__ cbuf,
                                                  float* __restrict__ out, int T) {
  long tok = blockIdx.x;
  int  t   = (int)(tok % T);
  long b   = tok / T;
  float c  = cbuf[tok];
  for (int d = threadIdx.x; d < D_MODEL; d += 256) {
    float conv = 0.f;
#pragma unroll
    for (int k = 0; k < 9; ++k) {
      int tt = t + k - 4;
      float xv = (tt >= 0 && tt < T) ? y[(b * T + tt) * D_MODEL + d] : 0.f;
      conv += dw[d * 9 + k] * xv;
    }
    float yv = y[tok * D_MODEL + d];
    out[tok * D_MODEL + d] = yv + (c * yv + (1.0f - c) * conv);
  }
}

// ---------------------------------------------------------------------------
// Fused band-masked flash attention, all-bf16 operands.  qkv: [B,T,1536] bf16
// (Q|K|V), head h owns 64-wide slices.  One wave per (b, h, 16-query tile).
// QK^T and PV via v_wmma; P transposed C-layout -> A-layout through LDS.
// ---------------------------------------------------------------------------
__global__ __launch_bounds__(32) void band_attn_kernel(
    const __bf16* __restrict__ qkv, __bf16* __restrict__ out, int band, int T) {
  __shared__ unsigned short pH[16 * 32];
  const int lane = threadIdx.x;
  const int half = lane >> 4;
  const int r    = lane & 15;
  const int t0 = blockIdx.x * 16;
  const int h  = blockIdx.y;
  const long b = blockIdx.z;

  const __bf16* base = qkv + b * T * 1536;
  const __bf16* Qb = base + h * HDIM;
  const __bf16* Kb = base + D_MODEL + h * HDIM;
  const __bf16* Vb = base + 2 * D_MODEL + h * HDIM;

  v16bf qf0 = load_frag_a_h(Qb, 1536, t0 + r, 0,  half);
  v16bf qf1 = load_frag_a_h(Qb, 1536, t0 + r, 32, half);

  float mrow[8], lrow[8];
  v8f   oacc[4];
#pragma unroll
  for (int rr = 0; rr < 8; ++rr) { mrow[rr] = 0.f; lrow[rr] = 0.f; }
#pragma unroll
  for (int nt = 0; nt < 4; ++nt)
#pragma unroll
    for (int i = 0; i < 8; ++i) oacc[nt][i] = 0.f;

  int s_begin = t0 - band; if (s_begin < 0) s_begin = 0;
  s_begin &= ~31;
  int s_end = t0 + 15 + band; if (s_end > T - 1) s_end = T - 1;

  for (int s0 = s_begin; s0 <= s_end; s0 += 32) {
    // ---- S = Q K^T : 16 queries x 32 keys, hd = 64 contraction ----
    v8f sc[2];
#pragma unroll
    for (int nt = 0; nt < 2; ++nt) {
#pragma unroll
      for (int i = 0; i < 8; ++i) sc[nt][i] = 0.f;
      int s   = s0 + nt * 16 + r;
      int scl = s < 0 ? 0 : (s > T - 1 ? T - 1 : s);
      v16bf kf0 = load_frag_bT_h(Kb, 1536, scl, 0,  half);
      v16bf kf1 = load_frag_bT_h(Kb, 1536, scl, 32, half);
      sc[nt] = wmma_bf16(qf0, kf0, sc[nt]);
      sc[nt] = wmma_bf16(qf1, kf1, sc[nt]);
    }
    // ---- mask + online softmax (row reductions across 16-lane halves) ----
    float pv[2][8], mnew[8];
#pragma unroll
    for (int rr = 0; rr < 8; ++rr) {
      int trow = t0 + half * 8 + rr;
      float loc = -1e30f;
#pragma unroll
      for (int nt = 0; nt < 2; ++nt) {
        int s = s0 + nt * 16 + r;
        float v = sc[nt][rr] * 0.125f;   // 1/sqrt(64)
        bool ok = (s >= 0) && (s < T) && (s >= trow - band) && (s <= trow + band);
        v = ok ? v : -1e30f;
        pv[nt][rr] = v;
        loc = fmaxf(loc, v);
      }
#pragma unroll
      for (int m = 1; m < 16; m <<= 1) loc = fmaxf(loc, __shfl_xor(loc, m, 16));
      float mn = fmaxf(mrow[rr], loc);
      mnew[rr] = mn;
      float alpha = __expf(mrow[rr] - mn);
      mrow[rr] = mn;
      lrow[rr] *= alpha;
#pragma unroll
      for (int nt = 0; nt < 4; ++nt) oacc[nt][rr] = oacc[nt][rr] * alpha;
    }
#pragma unroll
    for (int rr = 0; rr < 8; ++rr) {
      float rs = 0.f;
#pragma unroll
      for (int nt = 0; nt < 2; ++nt) {
        float p = __expf(pv[nt][rr] - mnew[rr]);
        rs += p;
        pH[(half * 8 + rr) * 32 + nt * 16 + r] = f32_bf16(p);  // C -> LDS (bf16)
      }
#pragma unroll
      for (int m = 1; m < 16; m <<= 1) rs += __shfl_xor(rs, m, 16);
      lrow[rr] += rs;
    }
    __syncthreads();
    // ---- P as A-fragment from LDS: two 16B ds vector loads ----
    v8bf plo = *(const v8bf*)(pH + r * 32 + 8 * half);
    v8bf phi = *(const v8bf*)(pH + r * 32 + 8 * half + 16);
    v16bf pf = __builtin_shufflevector(plo, phi, 0, 1, 2, 3, 4, 5, 6, 7,
                                       8, 9, 10, 11, 12, 13, 14, 15);
    // ---- O += P V   (B[k][n] = V[s0+k][n], strided rows -> scalar u16) ----
    const unsigned short* Vu = (const unsigned short*)Vb;
#pragma unroll
    for (int nt4 = 0; nt4 < 4; ++nt4) {
      v16us u;
#pragma unroll
      for (int j = 0; j < 8; ++j) {
        int k0 = 2 * j + 8 * half + (j >= 4 ? 8 : 0);
        int s1 = s0 + k0;     int s1c = s1 < 0 ? 0 : (s1 > T - 1 ? T - 1 : s1);
        int s2 = s0 + k0 + 1; int s2c = s2 < 0 ? 0 : (s2 > T - 1 ? T - 1 : s2);
        u[2 * j + 0] = Vu[(long)s1c * 1536 + nt4 * 16 + r];
        u[2 * j + 1] = Vu[(long)s2c * 1536 + nt4 * 16 + r];
      }
      v16bf vf = __builtin_bit_cast(v16bf, u);
      oacc[nt4] = wmma_bf16(pf, vf, oacc[nt4]);
    }
    __syncthreads();
  }
  // ---- epilogue: out[b, t, h*64 + n] = bf16(O / l) ----
  unsigned short* ou = (unsigned short*)out;
#pragma unroll
  for (int nt4 = 0; nt4 < 4; ++nt4) {
#pragma unroll
    for (int rr = 0; rr < 8; ++rr) {
      long trow = t0 + half * 8 + rr;
      ou[(b * T + trow) * D_MODEL + h * HDIM + nt4 * 16 + r] =
          f32_bf16(oacc[nt4][rr] / lrow[rr]);
    }
  }
}

// ---------------------------------------------------------------------------
// Host orchestration
// ---------------------------------------------------------------------------
extern "C" void kernel_launch(void* const* d_in, const int* in_sizes, int n_in,
                              void* d_out, int out_size, void* d_ws, size_t ws_size,
                              hipStream_t stream) {
  auto F = [&](int i) { return (const float*)d_in[i]; };
  const float* x_m      = F(0);
  const float* Ag       = F(1);
  const float* Hg       = F(2);
  const float* inp_b    = F(3);
  const float* inp_w    = F(4);
  const float* out_ln_b = F(55);
  const float* out_ln_g = F(56);

  const int M = B_BATCH * T_SEQ;   // 8192 rows

  // ---- workspace carve: f32 region first, then bf16 region ----
  float* wsf = (float*)d_ws;
  size_t foff = 0;
  auto allocF = [&](size_t n) { float* p = wsf + foff; foff += n; return p; };
  float* mem  = allocF((size_t)M * 512);
  float* y0   = allocF((size_t)M * 512);
  float* y1   = allocF((size_t)M * 512);
  float* cbuf = allocF((size_t)M);

  __bf16* wsh = (__bf16*)(wsf + foff);
  size_t hoff = 0;
  auto allocH = [&](size_t n) { __bf16* p = wsh + hoff; hoff += n; return p; };
  __bf16* xh    = allocH((size_t)M * 512);
  __bf16* tmpAh = allocH((size_t)M * 512);
  __bf16* tmpBh = allocH((size_t)M * 512);
  __bf16* qkvh  = allocH((size_t)M * 1536);
  __bf16* f1h   = allocH((size_t)M * 2048);
  __bf16* w_inp = allocH((size_t)512 * 512);
  __bf16* wbL[2][6];
  for (int L = 0; L < 2; ++L) {
    wbL[L][0] = allocH((size_t)1536 * 512);   // lsa_in
    wbL[L][1] = allocH((size_t)512 * 512);    // lsa_out
    wbL[L][2] = allocH((size_t)1536 * 512);   // rep_in
    wbL[L][3] = allocH((size_t)512 * 512);    // rep_out
    wbL[L][4] = allocH((size_t)2048 * 512);   // ffn_w1
    wbL[L][5] = allocH((size_t)512 * 2048);   // ffn_w2
  }

  auto cvt = [&](const float* src, __bf16* dst, long n) {
    long n8 = n / 8;
    cvt_bf16_kernel<<<dim3((unsigned)((n8 + 255) / 256)), 256, 0, stream>>>(src, dst, n8);
  };
  auto gemm = [&](const __bf16* A, int lda, const __bf16* W, const float* bias,
                  float* outf, __bf16* outh, int ldo,
                  const float* resid, int ldr, int N, int K, int act) {
    dim3 g(N / 256, M / 32);
    gemm_wmma_kernel<<<g, 256, 0, stream>>>(A, lda, W, bias, outf, outh, ldo,
                                            resid, ldr, M, N, K, act);
  };

  // ---- one-time conversions (deterministic, every call) ----
  cvt(x_m, xh, (long)M * 512);
  cvt(inp_w, w_inp, 512L * 512);
  for (int L = 0; L < 2; ++L) {
    const int pi = 5 + 25 * L;
    cvt(F(pi + 12), wbL[L][0], 1536L * 512);   // lsa.in_w
    cvt(F(pi + 14), wbL[L][1], 512L * 512);    // lsa.out_w
    cvt(F(pi + 18), wbL[L][2], 1536L * 512);   // rep.in_w
    cvt(F(pi + 20), wbL[L][3], 512L * 512);    // rep.out_w
    cvt(F(pi + 4),  wbL[L][4], 2048L * 512);   // ffn_w1
    cvt(F(pi + 5),  wbL[L][5], 512L * 2048);   // ffn_w2
  }

  // mem = inp(x_m); y = mem
  gemm(xh, 512, w_inp, inp_b, mem, nullptr, 512, nullptr, 0, 512, 512, 0);
  hipMemcpyAsync(y0, mem, (size_t)M * 512 * sizeof(float),
                 hipMemcpyDeviceToDevice, stream);
  float* y = y0;
  float* yalt = y1;

  for (int L = 0; L < 2; ++L) {
    const int pi = 5 + 25 * L;      // pytree-sorted leaf order within layer
    const float* ffn_b1    = F(pi + 0);
    const float* ffn_b2    = F(pi + 1);
    const float* ffn_ln_b  = F(pi + 2);
    const float* ffn_ln_g  = F(pi + 3);
    const float* g1_b      = F(pi + 7);
    const float* g1_w      = F(pi + 8);
    const float* g2_b      = F(pi + 9);
    const float* g2_w      = F(pi + 10);
    const float* dw_w      = F(pi + 6);
    const float* lsa_in_b  = F(pi + 11);
    const float* lsa_out_b = F(pi + 13);
    const float* lsa_ln_b  = F(pi + 15);
    const float* lsa_ln_g  = F(pi + 16);
    const float* rep_in_b  = F(pi + 17);
    const float* rep_out_b = F(pi + 19);
    const float* rep_lnk_b = F(pi + 21);
    const float* rep_lnk_g = F(pi + 22);
    const float* rep_lnq_b = F(pi + 23);
    const float* rep_lnq_g = F(pi + 24);

    // y = y + LPF(y)
    coeff_kernel<<<M / 8, 256, 0, stream>>>(Ag, Hg, g1_w, g1_b, g2_w, g2_b, cbuf, M);
    lpf_kernel<<<M, 256, 0, stream>>>(y, dw_w, cbuf, yalt, T_SEQ);
    { float* t = y; y = yalt; yalt = t; }

    // local self-attention (band 16)
    ln_kernel<<<M / 8, 256, 0, stream>>>(y, lsa_ln_g, lsa_ln_b, tmpAh, M);
    gemm(tmpAh, 512, wbL[L][0], lsa_in_b, nullptr, qkvh, 1536, nullptr, 0, 1536, 512, 0);
    band_attn_kernel<<<dim3(T_SEQ / 16, NHEADS, B_BATCH), 32, 0, stream>>>(
        qkvh, tmpBh, 16, T_SEQ);
    gemm(tmpBh, 512, wbL[L][1], lsa_out_b, y, nullptr, 512, y, 512, 512, 512, 0);

    // cross attention vs mem (band 24)
    ln_kernel<<<M / 8, 256, 0, stream>>>(y, rep_lnq_g, rep_lnq_b, tmpAh, M);
    ln_kernel<<<M / 8, 256, 0, stream>>>(mem, rep_lnk_g, rep_lnk_b, tmpBh, M);
    gemm(tmpAh, 512, wbL[L][2], rep_in_b, nullptr, qkvh, 1536, nullptr, 0, 512, 512, 0);
    gemm(tmpBh, 512, wbL[L][2] + 512 * 512, rep_in_b + 512, nullptr, qkvh + 512,
         1536, nullptr, 0, 1024, 512, 0);
    band_attn_kernel<<<dim3(T_SEQ / 16, NHEADS, B_BATCH), 32, 0, stream>>>(
        qkvh, tmpAh, 24, T_SEQ);
    gemm(tmpAh, 512, wbL[L][3], rep_out_b, y, nullptr, 512, y, 512, 512, 512, 0);

    // FFN
    ln_kernel<<<M / 8, 256, 0, stream>>>(y, ffn_ln_g, ffn_ln_b, tmpAh, M);
    gemm(tmpAh, 512, wbL[L][4], ffn_b1, nullptr, f1h, 2048, nullptr, 0, 2048, 512, 1);
    gemm(f1h, 2048, wbL[L][5], ffn_b2, y, nullptr, 512, y, 512, 512, 2048, 0);
  }

  // final LN must produce f32 into d_out: reuse ln math inline via a tiny
  // wrapper — write bf16 path is not suitable here, so do it with a dedicated
  // f32 LN (below).
  {
    // f32-output LayerNorm (separate kernel launch)
    extern __global__ void ln_f32_kernel(const float*, const float*, const float*,
                                         float*, int);
    ln_f32_kernel<<<M / 8, 256, 0, stream>>>(y, out_ln_g, out_ln_b,
                                             (float*)d_out, M);
  }
  (void)in_sizes; (void)n_in; (void)out_size; (void)ws_size;
}

// f32-output LayerNorm for the final projection to d_out.
__global__ __launch_bounds__(256) void ln_f32_kernel(const float* __restrict__ x,
                                                     const float* __restrict__ g,
                                                     const float* __restrict__ b,
                                                     float* __restrict__ o, int rows) {
  int row  = blockIdx.x * 8 + (threadIdx.x >> 5);
  int lane = threadIdx.x & 31;
  if (row >= rows) return;
  const float4* xr = (const float4*)(x + (long)row * D_MODEL);
  float4 v[4];
  float s = 0.f;
#pragma unroll
  for (int i = 0; i < 4; ++i) {
    v[i] = xr[lane + i * 32];
    s += v[i].x + v[i].y + v[i].z + v[i].w;
  }
#pragma unroll
  for (int m = 16; m; m >>= 1) s += __shfl_xor(s, m, 32);
  float mean = s * (1.0f / D_MODEL);
  float q = 0.f;
#pragma unroll
  for (int i = 0; i < 4; ++i) {
    float dx;
    dx = v[i].x - mean; q += dx * dx;
    dx = v[i].y - mean; q += dx * dx;
    dx = v[i].z - mean; q += dx * dx;
    dx = v[i].w - mean; q += dx * dx;
  }
#pragma unroll
  for (int m = 16; m; m >>= 1) q += __shfl_xor(q, m, 32);
  float inv = rsqrtf(q * (1.0f / D_MODEL) + 1e-5f);
  float4* orp = (float4*)(o + (long)row * D_MODEL);
#pragma unroll
  for (int i = 0; i < 4; ++i) {
    int e = (lane + i * 32) * 4;
    float4 w;
    w.x = (v[i].x - mean) * inv * g[e + 0] + b[e + 0];
    w.y = (v[i].y - mean) * inv * g[e + 1] + b[e + 1];
    w.z = (v[i].z - mean) * inv * g[e + 2] + b[e + 2];
    w.w = (v[i].w - mean) * inv * g[e + 3] + b[e + 3];
    orp[lane + i * 32] = w;
  }
}
